// TAGM_15247133901424
// MI455X (gfx1250) — matrix operand order repaired and checked by reading, not compiled, not run on hardware
//
#include <hip/hip_runtime.h>

typedef __bf16 bf16_t;
typedef __attribute__((ext_vector_type(16))) __bf16 v16bf;
typedef __attribute__((ext_vector_type(8)))  float  v8f;

// Problem constants
#define BB 64
#define TT 512
#define DD 256
#define HH 1024
#define CC 128

// Workspace layout (bytes)
static constexpr size_t SZ_WPACK_LSTM = 1280ull * 4096 * 2;  // bf16
static constexpr size_t SZ_WPACK_TAGM = 1280ull * 1024 * 2;  // bf16
static constexpr size_t OFF_WF   = 0;
static constexpr size_t OFF_WB   = OFF_WF + SZ_WPACK_LSTM;
static constexpr size_t OFF_WT   = OFF_WB + SZ_WPACK_LSTM;
static constexpr size_t OFF_ZERO = OFF_WT + SZ_WPACK_TAGM;          // zeroed from here
static constexpr size_t OFF_HL   = OFF_ZERO;                        // [2 dir][2 buf][64*1024] f32
static constexpr size_t OFF_CL   = OFF_HL + 4ull * 65536 * 4;       // [2 dir][64*1024] f32
static constexpr size_t OFF_HT   = OFF_CL + 2ull * 65536 * 4;       // [2 buf][64*1024] f32
static constexpr size_t OFF_SC   = OFF_HT + 2ull * 65536 * 4;       // [512*64] f32 scores
static constexpr size_t OFF_AT   = OFF_SC + 32768ull * 4;           // [512*64] f32 att gate
static constexpr size_t OFF_BAR  = OFF_AT + 32768ull * 4;           // barrier words
static constexpr size_t WS_TOTAL = OFF_BAR + 256;
static constexpr int    ZERO_N   = (int)((WS_TOTAL - OFF_ZERO) / 4);

__device__ __forceinline__ float sigm_f(float x) { return 1.f / (1.f + __expf(-x)); }

__device__ __forceinline__ void grid_barrier(unsigned* cnt, unsigned* gen, unsigned nwg) {
  __syncthreads();
  if (threadIdx.x == 0) {
    __threadfence();
    unsigned g = __atomic_load_n(gen, __ATOMIC_RELAXED);
    if (atomicAdd(cnt, 1u) == nwg - 1u) {
      __atomic_store_n(cnt, 0u, __ATOMIC_RELAXED);
      __atomic_fetch_add(gen, 1u, __ATOMIC_RELEASE);
    } else {
      while (__atomic_load_n(gen, __ATOMIC_ACQUIRE) == g) {
        __builtin_amdgcn_s_sleep(2);
      }
    }
    __threadfence();
  }
  __syncthreads();
}

// ---------------- init ----------------
__global__ void zero_f32(float* __restrict__ p, int n) {
  for (int i = blockIdx.x * 256 + threadIdx.x; i < n; i += gridDim.x * 256) p[i] = 0.f;
}

// ---------------- weight packing into WMMA B-fragment layout ----------------
// Augmented W = [Wh (1024 x N) ; Wi (256 x N)] -> K = 1280.
// packed[((kt*NT + nt)*32 + lane)*16 + j] = W[kt*32 + (lane>>4)*16 + j][nt*16 + (lane&15)]
__global__ void pack_w(const float* __restrict__ wh, const float* __restrict__ wi,
                       bf16_t* __restrict__ outp, int Ncols) {
  const int NT = Ncols >> 4;
  const long total = (long)1280 * Ncols;
  for (long idx = (long)blockIdx.x * 256 + threadIdx.x; idx < total;
       idx += (long)gridDim.x * 256) {
    int  j    = (int)(idx & 15);
    int  lane = (int)((idx >> 4) & 31);
    long rest = idx >> 9;
    int  nt   = (int)(rest % NT);
    int  kt   = (int)(rest / NT);
    int  K    = kt * 32 + ((lane >> 4) << 4) + j;
    int  n    = nt * 16 + (lane & 15);
    float v = (K < HH) ? wh[(long)K * Ncols + n] : wi[(long)(K - HH) * Ncols + n];
    outp[idx] = (bf16_t)v;
  }
}

// ---------------- bidirectional LSTM, persistent, both dirs concurrent ----------------
// grid = 32 WGs x 256 threads. dir = blockIdx.x>>4. 16 WGs per direction.
// Per direction: 128 waves; wave gw: jt = gw>>1 (hidden tile 0..63), mh = gw&1 (batch half).
__global__ __launch_bounds__(256) void lstm_kernel(
    const float* __restrict__ x,
    const bf16_t* __restrict__ wpackF, const bf16_t* __restrict__ wpackB,
    const float* __restrict__ bF, const float* __restrict__ bB,
    const float* __restrict__ attw,          // att_fc_w [2048]
    float* __restrict__ hbuf,                // [2 dir][2 buf][64*1024]
    float* __restrict__ cbuf,                // [2 dir][64*1024]
    float* __restrict__ score,               // [512*64]
    unsigned* __restrict__ bar) {
  __shared__ __align__(32) bf16_t smA[4 * 8 * 512];  // 32KB: 4 Mtiles x 8 Kchunks, A-frag layout

  const int dir  = blockIdx.x >> 4;
  const int tid  = threadIdx.x;
  const int lane = tid & 31;
  const int wv   = tid >> 5;
  const int gw   = (blockIdx.x & 15) * 8 + wv;  // 0..127
  const int jt   = gw >> 1;                     // hidden tile
  const int mh   = gw & 1;                      // batch half (rows mh*32..)

  const bf16_t* wpack = dir ? wpackB : wpackF;
  const float*  b4h   = dir ? bB : bF;
  float* hB = hbuf + (size_t)dir * 2 * 65536;
  float* cB = cbuf + (size_t)dir * 65536;
  unsigned* cnt = bar + dir;
  unsigned* gen = bar + 2 + dir;

  const int   colg = jt * 16 + (lane & 15);
  const float bi  = b4h[colg];
  const float bfv = b4h[HH + colg];
  const float bgv = b4h[2 * HH + colg];
  const float bov = b4h[3 * HH + colg];
  const float wsc = attw[dir * HH + colg];

  for (int step = 0; step < TT; ++step) {
    const int time = dir ? (TT - 1 - step) : step;
    const float* hin  = hB + (step & 1) * 65536;
    float*       hout = hB + ((step + 1) & 1) * 65536;

    v8f acc[2][4];
    #pragma unroll
    for (int m = 0; m < 2; ++m)
      #pragma unroll
      for (int g = 0; g < 4; ++g) {
        v8f z = {0.f, 0.f, 0.f, 0.f, 0.f, 0.f, 0.f, 0.f};
        acc[m][g] = z;
      }

    for (int kg = 0; kg < 5; ++kg) {  // 5 groups x 8 Kchunks x 32 = K 1280
      // Cooperative stage of A = [h | x_t] (bf16, A-fragment layout) into LDS
      for (int idx = tid; idx < 16384; idx += 256) {
        int j    = idx & 15;
        int ls   = (idx >> 4) & 31;
        int frag = idx >> 9;     // mt*8 + kc
        int kc   = frag & 7;
        int mt   = frag >> 3;
        int s    = ls >> 4;
        int m    = ls & 15;
        int kl   = j + 16 * s;
        int K    = kg * 256 + kc * 32 + kl;
        int row  = mt * 16 + m;
        float v  = (K < HH) ? hin[row * HH + K]
                            : x[((long)row * TT + time) * DD + (K - HH)];
        smA[idx] = (bf16_t)v;
      }
      __syncthreads();

      const v16bf* av = (const v16bf*)smA;
      const v16bf* wv4 = (const v16bf*)wpack;
      #pragma unroll 2
      for (int kc = 0; kc < 8; ++kc) {
        const int kt = kg * 8 + kc;
        v16bf a0 = av[((mh * 2 + 0) * 8 + kc) * 32 + lane];
        v16bf a1 = av[((mh * 2 + 1) * 8 + kc) * 32 + lane];
        #pragma unroll
        for (int g = 0; g < 4; ++g) {
          v16bf bfrag = wv4[((size_t)kt * 256 + g * 64 + jt) * 32 + lane];
          acc[0][g] = __builtin_amdgcn_wmma_f32_16x16x32_bf16(
              false, a0, false, bfrag, (short)0, acc[0][g], false, false);
          acc[1][g] = __builtin_amdgcn_wmma_f32_16x16x32_bf16(
              false, a1, false, bfrag, (short)0, acc[1][g], false, false);
        }
      }
      __syncthreads();
    }

    // LSTM cell update + attention-score contribution
    #pragma unroll
    for (int mt = 0; mt < 2; ++mt) {
      #pragma unroll
      for (int r = 0; r < 8; ++r) {
        int row = (mh * 2 + mt) * 16 + r + ((lane >> 4) << 3);
        float iv = sigm_f(acc[mt][0][r] + bi);
        float fv = sigm_f(acc[mt][1][r] + bfv);
        float gv = tanhf(acc[mt][2][r] + bgv);
        float ov = sigm_f(acc[mt][3][r] + bov);
        float cold = cB[row * HH + colg];
        float cnew = fv * cold + iv * gv;
        float hnew = ov * tanhf(cnew);
        cB[row * HH + colg]   = cnew;
        hout[row * HH + colg] = hnew;
        // reduce h*w over the 16 hidden columns held by this half-wave
        float v = hnew * wsc;
        v += __shfl_xor(v, 1, 32);
        v += __shfl_xor(v, 2, 32);
        v += __shfl_xor(v, 4, 32);
        v += __shfl_xor(v, 8, 32);
        if ((lane & 15) == 0) atomicAdd(&score[time * BB + row], v);
      }
    }
    grid_barrier(cnt, gen, 16);
  }
}

// ---------------- attention gate ----------------
__global__ void att_fin(const float* __restrict__ score, const float* __restrict__ attb,
                        float* __restrict__ att, float* __restrict__ out_att) {
  int idx = blockIdx.x * 256 + threadIdx.x;
  if (idx < TT * BB) {
    int t = idx >> 6, b = idx & 63;
    float a = 1.f / (1.f + __expf(-3.f * (score[idx] + attb[0])));
    att[idx] = a;
    out_att[b * TT + t] = a;  // attention output is [B,T]
  }
}

// ---------------- TAGM recurrence, persistent ----------------
// grid = 16 WGs x 256. 128 waves: jt = gw>>1 (0..63), mh = gw&1.
__global__ __launch_bounds__(256) void tagm_kernel(
    const float* __restrict__ x,
    const bf16_t* __restrict__ wpack,        // [1280 x 1024] packed
    const float* __restrict__ i2hb, const float* __restrict__ h2hb,
    const float* __restrict__ att,           // [512*64]
    float* __restrict__ hbuf,                // [2 buf][64*1024]
    unsigned* __restrict__ bar) {
  __shared__ __align__(32) bf16_t smA[4 * 8 * 512];

  const int tid  = threadIdx.x;
  const int lane = tid & 31;
  const int wv   = tid >> 5;
  const int gw   = blockIdx.x * 8 + wv;  // 0..127
  const int jt   = gw >> 1;
  const int mh   = gw & 1;
  unsigned* cnt = bar + 4;
  unsigned* gen = bar + 5;

  const int   colg = jt * 16 + (lane & 15);
  const float bb   = i2hb[colg] + h2hb[colg];

  for (int step = 0; step < TT; ++step) {
    const float* hin  = hbuf + (step & 1) * 65536;
    float*       hout = hbuf + ((step + 1) & 1) * 65536;

    v8f acc[2];
    {
      v8f z = {0.f, 0.f, 0.f, 0.f, 0.f, 0.f, 0.f, 0.f};
      acc[0] = z; acc[1] = z;
    }

    for (int kg = 0; kg < 5; ++kg) {
      for (int idx = tid; idx < 16384; idx += 256) {
        int j    = idx & 15;
        int ls   = (idx >> 4) & 31;
        int frag = idx >> 9;
        int kc   = frag & 7;
        int mt   = frag >> 3;
        int s    = ls >> 4;
        int m    = ls & 15;
        int kl   = j + 16 * s;
        int K    = kg * 256 + kc * 32 + kl;
        int row  = mt * 16 + m;
        float v  = (K < HH) ? hin[row * HH + K]
                            : x[((long)row * TT + step) * DD + (K - HH)];
        smA[idx] = (bf16_t)v;
      }
      __syncthreads();

      const v16bf* av = (const v16bf*)smA;
      const v16bf* wv4 = (const v16bf*)wpack;
      #pragma unroll 4
      for (int kc = 0; kc < 8; ++kc) {
        const int kt = kg * 8 + kc;
        v16bf a0 = av[((mh * 2 + 0) * 8 + kc) * 32 + lane];
        v16bf a1 = av[((mh * 2 + 1) * 8 + kc) * 32 + lane];
        v16bf bfrag = wv4[((size_t)kt * 64 + jt) * 32 + lane];
        acc[0] = __builtin_amdgcn_wmma_f32_16x16x32_bf16(
            false, a0, false, bfrag, (short)0, acc[0], false, false);
        acc[1] = __builtin_amdgcn_wmma_f32_16x16x32_bf16(
            false, a1, false, bfrag, (short)0, acc[1], false, false);
      }
      __syncthreads();
    }

    #pragma unroll
    for (int mt = 0; mt < 2; ++mt) {
      #pragma unroll
      for (int r = 0; r < 8; ++r) {
        int row = (mh * 2 + mt) * 16 + r + ((lane >> 4) << 3);
        float cand = acc[mt][r] + bb;
        cand = fmaxf(cand, 0.f);
        float a    = att[step * BB + row];
        float hold = hin[row * HH + colg];
        hout[row * HH + colg] = fmaf(a, cand - hold, hold);  // a*cand + (1-a)*hold
      }
    }
    grid_barrier(cnt, gen, 16);
  }
}

// ---------------- final FC: [64x1024] @ [1024x128] + b ----------------
__global__ void fc_kernel(const float* __restrict__ h, const float* __restrict__ w,
                          const float* __restrict__ b, float* __restrict__ out) {
  int idx = blockIdx.x * 256 + threadIdx.x;
  if (idx < BB * CC) {
    int bb = idx >> 7, cc = idx & 127;
    float s = b[cc];
    const float* hr = h + bb * HH;
    #pragma unroll 4
    for (int k = 0; k < HH; ++k) s += hr[k] * w[k * CC + cc];
    out[idx] = s;
  }
}

extern "C" void kernel_launch(void* const* d_in, const int* in_sizes, int n_in,
                              void* d_out, int out_size, void* d_ws, size_t ws_size,
                              hipStream_t stream) {
  const float* x          = (const float*)d_in[0];
  const float* i2h_w      = (const float*)d_in[1];
  const float* i2h_b      = (const float*)d_in[2];
  const float* h2h_w      = (const float*)d_in[3];
  const float* h2h_b      = (const float*)d_in[4];
  const float* fc_w       = (const float*)d_in[5];
  const float* fc_b       = (const float*)d_in[6];
  const float* att_wi_fwd = (const float*)d_in[7];
  const float* att_wh_fwd = (const float*)d_in[8];
  const float* att_b_fwd  = (const float*)d_in[9];
  const float* att_wi_bwd = (const float*)d_in[10];
  const float* att_wh_bwd = (const float*)d_in[11];
  const float* att_b_bwd  = (const float*)d_in[12];
  const float* att_fc_w   = (const float*)d_in[13];
  const float* att_fc_b   = (const float*)d_in[14];

  char* ws = (char*)d_ws;
  bf16_t*   wf    = (bf16_t*)(ws + OFF_WF);
  bf16_t*   wb    = (bf16_t*)(ws + OFF_WB);
  bf16_t*   wt    = (bf16_t*)(ws + OFF_WT);
  float*    hl    = (float*)(ws + OFF_HL);
  float*    cl    = (float*)(ws + OFF_CL);
  float*    ht    = (float*)(ws + OFF_HT);
  float*    sc    = (float*)(ws + OFF_SC);
  float*    at    = (float*)(ws + OFF_AT);
  unsigned* bar   = (unsigned*)(ws + OFF_BAR);

  float* out_fc  = (float*)d_out;            // [64*128]
  float* out_att = (float*)d_out + BB * CC;  // [64*512]

  // 1. zero states / scores / barriers
  zero_f32<<<256, 256, 0, stream>>>((float*)(ws + OFF_ZERO), ZERO_N);
  // 2. pack weights to bf16 WMMA fragments (L2-resident thereafter: ~23MB << 192MB L2)
  pack_w<<<2048, 256, 0, stream>>>(att_wh_fwd, att_wi_fwd, wf, 4 * HH);
  pack_w<<<2048, 256, 0, stream>>>(att_wh_bwd, att_wi_bwd, wb, 4 * HH);
  pack_w<<<512, 256, 0, stream>>>(h2h_w, i2h_w, wt, HH);
  // 3. bidirectional LSTM (persistent, both directions concurrent)
  lstm_kernel<<<32, 256, 0, stream>>>(x, wf, wb, att_b_fwd, att_b_bwd, att_fc_w,
                                      hl, cl, sc, bar);
  // 4. attention gate
  att_fin<<<(TT * BB + 255) / 256, 256, 0, stream>>>(sc, att_fc_b, at, out_att);
  // 5. TAGM recurrence (persistent)
  tagm_kernel<<<16, 256, 0, stream>>>(x, wt, i2h_b, h2h_b, at, ht, bar);
  // 6. final FC (final hidden state lives in buffer 0 after 512 steps)
  fc_kernel<<<(BB * CC + 255) / 256, 256, 0, stream>>>(ht, fc_w, fc_b, out_fc);
}
